// GraphConv_12154757448184
// MI455X (gfx1250) — matrix operand order repaired and checked by reading, not compiled, not run on hardware
//
#include <hip/hip_runtime.h>
#include <hip/hip_bf16.h>
#include <cstdint>
#include <cstddef>

#define DEVI __device__ __forceinline__

typedef __attribute__((ext_vector_type(16))) __bf16 bf16x16;
typedef __attribute__((ext_vector_type(8)))  __bf16 bf16x8;
typedef __attribute__((ext_vector_type(8)))  float  f32x8;
typedef __attribute__((ext_vector_type(4)))  unsigned u32x4;
typedef __attribute__((ext_vector_type(8)))  int      i32x8;
typedef __attribute__((ext_vector_type(4)))  int      i32x4;

#ifndef __has_builtin
#define __has_builtin(x) 0
#endif
#if __has_builtin(__builtin_amdgcn_tensor_load_to_lds)
#define HAS_TDM 1
#else
#define HAS_TDM 0
#endif
#if __has_builtin(__builtin_amdgcn_s_wait_tensorcnt)
#define WAIT_TENSORCNT0() __builtin_amdgcn_s_wait_tensorcnt(0)
#else
#define WAIT_TENSORCNT0() asm volatile("s_wait_tensorcnt 0" ::: "memory")
#endif
#if __has_builtin(__builtin_amdgcn_s_wait_asynccnt)
#define WAIT_ASYNCCNT0() __builtin_amdgcn_s_wait_asynccnt(0)
#else
#define WAIT_ASYNCCNT0() asm volatile("s_wait_asynccnt 0" ::: "memory")
#endif

constexpr int N_ = 65436;     // B*S - 100
constexpr int C_ = 256;
constexpr int B_ = 256;
constexpr int H_ = 8;
constexpr int E_ = 1000000;
constexpr float EPS_ = 1e-5f;
constexpr float QSCALE_ = 0.17677669529663687f;   // 1/sqrt(32)

// low 32 bits of a generic LDS address == LDS byte offset (aperture rule)
DEVI unsigned lds_off(const void* p) { return (unsigned)(uintptr_t)p; }

// async copy 16B global -> LDS (per-lane addresses), tracked by ASYNCcnt
DEVI void async_g2l_b128(unsigned lds_byte_off, const void* gsrc) {
  unsigned long long ga = (unsigned long long)(uintptr_t)gsrc;
  asm volatile("global_load_async_to_lds_b128 %0, %1, off"
               :: "v"(lds_byte_off), "v"(ga) : "memory");
}

// ---------------------------------------------------------------- WMMA core
DEVI f32x8 wmma_bf16(bf16x16 a, bf16x16 b, f32x8 c) {
  // v_wmma_f32_16x16x32_bf16: D = A(16x32) * B(32x16) + C
  return __builtin_amdgcn_wmma_f32_16x16x32_bf16(false, a, false, b,
                                                 (short)0, c, false, false);
}

// A fragment (16x32 bf16) from LDS tile, row-major with stride lda.
DEVI bf16x16 load_A_lds(const __bf16* base, int lda) {
  int lane = threadIdx.x & 31;
  const __bf16* p = base + (lane & 15) * lda + ((lane >> 4) << 4);
  bf16x8 lo = *(const bf16x8*)p;
  bf16x8 hi = *(const bf16x8*)(p + 8);
  bf16x16 r;
#pragma unroll
  for (int i = 0; i < 8; ++i) { r[i] = lo[i]; r[8 + i] = hi[i]; }
  return r;
}

// B fragment (32x16) = W^T tile, W row-major [COUT][ldk] bf16.
DEVI bf16x16 load_B_w(const __bf16* W, int ldk, int n0, int k0) {
  int lane = threadIdx.x & 31;
  const __bf16* p = W + (size_t)(n0 + (lane & 15)) * ldk + k0 + ((lane >> 4) << 4);
  return *(const bf16x16*)p;
}

// ---------------------------------------------------------------- small utils
__global__ void k_cast(const float* __restrict__ s, __bf16* __restrict__ d, int n) {
  int i = blockIdx.x * 256 + threadIdx.x;
  if (i < n) d[i] = (__bf16)s[i];
}

__global__ void k_count(const int* __restrict__ batch, int* __restrict__ counts, int n) {
  int i = blockIdx.x * 256 + threadIdx.x;
  if (i < n) atomicAdd(&counts[batch[i]], 1);
}

__global__ void k_scan(const int* __restrict__ counts, int* __restrict__ starts) {
  if (threadIdx.x == 0) {
    int acc = 0;
    for (int b = 0; b < B_; ++b) { starts[b] = acc; acc += counts[b]; }
  }
}

// ------------------------------------------------ edge scatter (bandwidth bound)
__global__ __launch_bounds__(256) void k_edge(const int* __restrict__ ei,
                                              const float* __restrict__ x,
                                              float* __restrict__ agg, int ne) {
  int t = blockIdx.x * 256 + threadIdx.x;
  int e = t >> 6;                 // 64 threads per edge, 4 ch each
  if (e >= ne) return;
  int c = (t & 63) << 2;
  int src = ei[e];
  int dst = ei[ne + e];
  float4 vx = *(const float4*)(x + (size_t)src * C_ + c);
  float* p = agg + (size_t)dst * C_ + c;
  atomicAdd(p + 0, vx.x);
  atomicAdd(p + 1, vx.y);
  atomicAdd(p + 2, vx.z);
  atomicAdd(p + 3, vx.w);
}

// ---------------------------------------- GraphConv + residual + BN1 (fused)
__global__ __launch_bounds__(256) void k_conv(
    const float* __restrict__ x, const float* __restrict__ agg,
    const __bf16* __restrict__ Wr, const __bf16* __restrict__ Wn,
    const float* __restrict__ bnei,
    const float* __restrict__ g1, const float* __restrict__ be1,
    const float* __restrict__ m1, const float* __restrict__ v1,
    float* __restrict__ h1, int Nn) {
  const int R = 32, LDA = 264;                 // +8 bf16 pad: conflict-free A reads
  __shared__ __bf16 sx[R * LDA];
  __shared__ __bf16 sg[R * LDA];
  int row0 = blockIdx.x * R;
  for (int i = threadIdx.x; i < R * 64; i += 256) {
    int r = i >> 6, c4 = (i & 63) << 2;
    int row = row0 + r;
    float4 qx = make_float4(0.f, 0.f, 0.f, 0.f);
    float4 qa = make_float4(0.f, 0.f, 0.f, 0.f);
    if (row < Nn) {
      qx = *(const float4*)(x + (size_t)row * C_ + c4);
      qa = *(const float4*)(agg + (size_t)row * C_ + c4);
    }
    __bf16* px = sx + r * LDA + c4;
    px[0] = (__bf16)qx.x; px[1] = (__bf16)qx.y; px[2] = (__bf16)qx.z; px[3] = (__bf16)qx.w;
    __bf16* pg = sg + r * LDA + c4;
    pg[0] = (__bf16)qa.x; pg[1] = (__bf16)qa.y; pg[2] = (__bf16)qa.z; pg[3] = (__bf16)qa.w;
  }
  __syncthreads();
  int w = threadIdx.x >> 5, lane = threadIdx.x & 31;
  for (int t = w; t < 32; t += 8) {            // 2 m-tiles x 16 n-tiles
    int mt = t & 1, nt = t >> 1;
    f32x8 acc = {};
#pragma unroll
    for (int k0 = 0; k0 < C_; k0 += 32) {
      bf16x16 ax = load_A_lds(sx + mt * 16 * LDA + k0, LDA);
      acc = wmma_bf16(ax, load_B_w(Wr, C_, nt * 16, k0), acc);
      bf16x16 ag = load_A_lds(sg + mt * 16 * LDA + k0, LDA);
      acc = wmma_bf16(ag, load_B_w(Wn, C_, nt * 16, k0), acc);
    }
    int n = nt * 16 + (lane & 15);
    int mr = mt * 16 + ((lane >> 4) << 3);
    float sc = g1[n] * rsqrtf(v1[n] + EPS_);
    float mu = m1[n], bt = be1[n], bb = bnei[n];
#pragma unroll
    for (int i = 0; i < 8; ++i) {
      int row = row0 + mr + i;
      if (row < Nn) {
        float tv = acc[i] + bb + x[(size_t)row * C_ + n];   // h + x (residual)
        h1[(size_t)row * C_ + n] = (tv - mu) * sc + bt;
      }
    }
  }
}

// --------------------------------- generic staged GEMM (CIN=256) with epilogue
template <int COUT, class Epi>
__global__ __launch_bounds__(256) void k_gemm(const float* __restrict__ A,
                                              const __bf16* __restrict__ W,
                                              Epi epi, int Nn) {
  const int R = 64, LDA = 264;
  __shared__ __bf16 sA[R * LDA];
  int row0 = blockIdx.x * R;
  if (threadIdx.x == 0) __builtin_prefetch(W, 0, 1);   // global_prefetch_b8
  for (int i = threadIdx.x; i < R * 64; i += 256) {
    int r = i >> 6, c4 = (i & 63) << 2;
    int row = row0 + r;
    float4 q = make_float4(0.f, 0.f, 0.f, 0.f);
    if (row < Nn) q = *(const float4*)(A + (size_t)row * C_ + c4);
    __bf16* p = sA + r * LDA + c4;
    p[0] = (__bf16)q.x; p[1] = (__bf16)q.y; p[2] = (__bf16)q.z; p[3] = (__bf16)q.w;
  }
  __syncthreads();
  int w = threadIdx.x >> 5, lane = threadIdx.x & 31;
  const int T = 4 * (COUT / 16);
  for (int t = w; t < T; t += 8) {
    int mt = t & 3, nt = t >> 2;
    f32x8 acc = {};
#pragma unroll
    for (int k0 = 0; k0 < C_; k0 += 32) {
      bf16x16 a = load_A_lds(sA + mt * 16 * LDA + k0, LDA);
      acc = wmma_bf16(a, load_B_w(W, C_, nt * 16, k0), acc);
    }
    int n = nt * 16 + (lane & 15);
    int mr = mt * 16 + ((lane >> 4) << 3);
#pragma unroll
    for (int i = 0; i < 8; ++i) {
      int row = row0 + mr + i;
      if (row < Nn) epi(row, n, acc[i]);
    }
  }
}

struct EpiQKV {     // qkv = x@Wqkv^T + bqkv; write q(pre-scaled)/k/v as bf16
  const float* bqkv; __bf16 *qo, *ko, *vo;
  __device__ void operator()(int row, int n, float val) const {
    val += bqkv[n];
    if (n < 256)       qo[(size_t)row * C_ + n]        = (__bf16)(val * QSCALE_);
    else if (n < 512)  ko[(size_t)row * C_ + (n - 256)] = (__bf16)val;
    else               vo[(size_t)row * C_ + (n - 512)] = (__bf16)val;
  }
};
struct EpiOproj {   // h2 = BN2(attn@Wo^T + bo + x); out = h1 + h2
  const float *bo, *x, *g, *be, *mm, *vv, *h1; float* outb;
  __device__ void operator()(int row, int n, float val) const {
    float t = val + bo[n] + x[(size_t)row * C_ + n];
    float h2 = (t - mm[n]) * (g[n] * rsqrtf(vv[n] + EPS_)) + be[n];
    outb[(size_t)row * C_ + n] = h1[(size_t)row * C_ + n] + h2;
  }
};
struct EpiMlp1 {    // t = relu(out@W1^T + b1) as bf16
  const float* b1; __bf16* t;
  __device__ void operator()(int row, int n, float val) const {
    t[(size_t)row * 512 + n] = (__bf16)fmaxf(val + b1[n], 0.f);
  }
};

// ---------------------------------------------- MLP second GEMM (CIN=512) + BN3
__global__ __launch_bounds__(256) void k_mlp2(
    const __bf16* __restrict__ T, const __bf16* __restrict__ W2,
    const float* __restrict__ b2, const float* __restrict__ outb,
    const float* __restrict__ g3, const float* __restrict__ be3,
    const float* __restrict__ m3, const float* __restrict__ v3,
    float* __restrict__ out, int Nn) {
  int row0 = blockIdx.x * 64;
  int w = threadIdx.x >> 5, lane = threadIdx.x & 31;
  for (int t = w; t < 64; t += 8) {
    int mt = t & 3, nt = t >> 2;
    int rowa = row0 + mt * 16 + (lane & 15);
    if (rowa > Nn - 1) rowa = Nn - 1;
    const __bf16* arow = T + (size_t)rowa * 512 + ((lane >> 4) << 4);
    f32x8 acc = {};
#pragma unroll
    for (int k0 = 0; k0 < 512; k0 += 32) {
      bf16x8 lo = *(const bf16x8*)(arow + k0);
      bf16x8 hi = *(const bf16x8*)(arow + k0 + 8);
      bf16x16 a;
#pragma unroll
      for (int i = 0; i < 8; ++i) { a[i] = lo[i]; a[8 + i] = hi[i]; }
      acc = wmma_bf16(a, load_B_w(W2, 512, nt * 16, k0), acc);
    }
    int n = nt * 16 + (lane & 15);
    int mr = mt * 16 + ((lane >> 4) << 3);
    float sc = g3[n] * rsqrtf(v3[n] + EPS_);
    float mu = m3[n], bt = be3[n], bv = b2[n];
#pragma unroll
    for (int i = 0; i < 8; ++i) {
      int row = row0 + mr + i;
      if (row < Nn) {
        float o = outb[(size_t)row * C_ + n];
        out[(size_t)row * C_ + n] = ((o + acc[i] + bv) - mu) * sc + bt;
      }
    }
  }
}

// ------------------------------------------------------- flash attention per (b,h)
DEVI bf16x16 frag_zero16() {
  bf16x16 r;
#pragma unroll
  for (int i = 0; i < 16; ++i) r[i] = (__bf16)0.f;
  return r;
}

DEVI bf16x16 load_Q(const __bf16* qb, int st, int cnt, int h, int q0, int lane) {
  int m = lane & 15, kb = (lane >> 4) << 4;
  int s = q0 + m;
  bf16x16 r = frag_zero16();
  if (s < cnt) {
    const __bf16* p = qb + (size_t)(st + s) * C_ + h * 32;
    bf16x8 lo = *(const bf16x8*)(p + kb);
    bf16x8 hi = *(const bf16x8*)(p + 8 + kb);
#pragma unroll
    for (int i = 0; i < 8; ++i) { r[i] = lo[i]; r[8 + i] = hi[i]; }
  }
  return r;
}

DEVI bf16x16 load_Bk(const __bf16* Ks, int key0, int lane) {   // B[k=d, n=key]
  const __bf16* p = Ks + (key0 + (lane & 15)) * 40 + ((lane >> 4) << 4);
  bf16x8 lo = *(const bf16x8*)p;
  bf16x8 hi = *(const bf16x8*)(p + 8);
  bf16x16 r;
#pragma unroll
  for (int i = 0; i < 8; ++i) { r[i] = lo[i]; r[8 + i] = hi[i]; }
  return r;
}

DEVI bf16x16 load_Bv(const __bf16* Vt, int key0, int dhalf, int lane) { // B[k=key, n=d]
  const __bf16* p = Vt + (dhalf + (lane & 15)) * 264 + key0 + ((lane >> 4) << 4);
  bf16x8 lo = *(const bf16x8*)p;
  bf16x8 hi = *(const bf16x8*)(p + 8);
  bf16x16 r;
#pragma unroll
  for (int i = 0; i < 8; ++i) { r[i] = lo[i]; r[8 + i] = hi[i]; }
  return r;
}

// Online-softmax update for one 16-row group; sa/sb hold 32 key columns.
DEVI void online_update(f32x8& mst, f32x8& lst, f32x8& sa, f32x8& sb,
                        f32x8& oa, f32x8& ob, __bf16* Pg, int lane) {
  f32x8 mx;
#pragma unroll
  for (int i = 0; i < 8; ++i) mx[i] = fmaxf(sa[i], sb[i]);
#pragma unroll
  for (int off = 1; off < 16; off <<= 1)
#pragma unroll
    for (int i = 0; i < 8; ++i) mx[i] = fmaxf(mx[i], __shfl_xor(mx[i], off, 32));
  f32x8 rs;
#pragma unroll
  for (int i = 0; i < 8; ++i) {
    float mn = fmaxf(mst[i], mx[i]);
    float sc = __expf(mst[i] - mn);
    float pa = __expf(sa[i] - mn);
    float pb = __expf(sb[i] - mn);
    sa[i] = pa; sb[i] = pb;
    rs[i] = pa + pb;
    lst[i] *= sc;
    oa[i] *= sc; ob[i] *= sc;
    mst[i] = mn;
  }
#pragma unroll
  for (int off = 1; off < 16; off <<= 1)
#pragma unroll
    for (int i = 0; i < 8; ++i) rs[i] += __shfl_xor(rs[i], off, 32);
#pragma unroll
  for (int i = 0; i < 8; ++i) lst[i] += rs[i];
  int col = lane & 15;
  int r0 = (lane >> 4) << 3;
#pragma unroll
  for (int i = 0; i < 8; ++i) {                 // transpose D-layout -> LDS rows
    Pg[(r0 + i) * 40 + col]      = (__bf16)sa[i];
    Pg[(r0 + i) * 40 + col + 16] = (__bf16)sb[i];
  }
}

DEVI void store_O(float* attn, int st, int cnt, int h, int qbase,
                  const f32x8& od0, const f32x8& od1, const f32x8& l, int lane) {
  int d = lane & 15;
  int r0 = (lane >> 4) << 3;
#pragma unroll
  for (int i = 0; i < 8; ++i) {
    int s = qbase + r0 + i;
    if (s < cnt) {
      float inv = 1.f / l[i];
      float* p = attn + (size_t)(st + s) * C_ + h * 32;
      p[d] = od0[i] * inv;
      p[16 + d] = od1[i] * inv;
    }
  }
}

__global__ __launch_bounds__(256) void k_attn(
    const __bf16* __restrict__ qb, const __bf16* __restrict__ kb,
    const __bf16* __restrict__ vb, const int* __restrict__ counts,
    const int* __restrict__ starts, float* __restrict__ attn) {
  __shared__ __bf16 Ks[256 * 40];      // [key][d], 80B padded rows
  __shared__ __bf16 Vt[32 * 264];      // [d][key]
  __shared__ __bf16 Ps[8 * 32 * 40];   // per-wave P transpose tile
  int b = blockIdx.x >> 3, h = blockIdx.x & 7;
  int cnt = counts[b], st = starts[b];

  // ---- K tile: Tensor Data Mover (zero-fills rows >= cnt, pads LDS rows to 80B)
#if HAS_TDM
  if (threadIdx.x < 32) {              // one TDM op, issued by wave 0
    unsigned long long ga = (unsigned long long)(uintptr_t)kb +
                            (unsigned long long)st * (C_ * 2) + (unsigned)h * 64u;
    u32x4 g0 = {};
    g0[0] = 1u;                                            // count = 1 (valid D#)
    g0[1] = lds_off(Ks);                                   // lds_addr (bytes)
    g0[2] = (unsigned)(ga & 0xFFFFFFFFull);                // global_addr[31:0]
    g0[3] = (unsigned)((ga >> 32) & 0x01FFFFFFull) | (2u << 30);  // addr[56:32] | type=2
    i32x8 g1 = {};
    g1[0] = (1 << 16)        // data_size = 2B
          | (1 << 20)        // pad_enable
          | (3 << 22)        // pad_interval = 16 DWORDs (one 64B row)
          | (3 << 25);       // pad_amount   = 4 DWORDs (16B) -> 80B LDS stride
    g1[1] = (32 & 0xFFFF) << 16;                           // tensor_dim0 = 32 elems
    g1[2] = (cnt & 0xFFFF) << 16;                          // tensor_dim1 = cnt (rows)
    g1[3] = ((cnt >> 16) & 0xFFFF) | (32 << 16);           // | tile_dim0 = 32
    g1[4] = 256;                                           // tile_dim1 = 256 rows
    g1[5] = 256;                                           // dim0_stride = 256 elems (512B)
    g1[6] = 0; g1[7] = 0;
    i32x4 zz = {};
#if defined(__clang_major__) && (__clang_major__ >= 23)
    i32x8 z8 = {};
    __builtin_amdgcn_tensor_load_to_lds(g0, g1, zz, zz, z8, 0);
#else
    __builtin_amdgcn_tensor_load_to_lds(g0, g1, zz, zz, 0);
#endif
    WAIT_TENSORCNT0();
  }
#else
  for (int i = threadIdx.x; i < 1024; i += 256) {
    int s = i >> 2, dc = (i & 3) << 3;
    bf16x8 kv;
    if (s < cnt) {
      kv = *(const bf16x8*)(kb + (size_t)(st + s) * C_ + h * 32 + dc);
    } else {
#pragma unroll
      for (int j = 0; j < 8; ++j) kv[j] = (__bf16)0.f;
    }
#pragma unroll
    for (int j = 0; j < 8; ++j) Ks[s * 40 + dc + j] = kv[j];
  }
#endif

  // ---- V tile: async global->LDS copy (ASYNCcnt), then LDS transpose to [d][key]
  {
    __shared__ __bf16 Vs[256 * 40];
    for (int i = threadIdx.x; i < 1024; i += 256) {
      int s = i >> 2, dc = (i & 3) << 3;
      const __bf16* gsrc = vb + (size_t)(st + s) * C_ + h * 32 + dc;
      async_g2l_b128(lds_off(Vs + s * 40 + dc), gsrc);
    }
    WAIT_ASYNCCNT0();
    __syncthreads();
    for (int i = threadIdx.x; i < 1024; i += 256) {
      int s = i >> 2, dc = (i & 3) << 3;
      bf16x8 vv = *(const bf16x8*)(Vs + s * 40 + dc);
      if (s >= cnt) {
#pragma unroll
        for (int j = 0; j < 8; ++j) vv[j] = (__bf16)0.f;
      }
#pragma unroll
      for (int j = 0; j < 8; ++j) Vt[(dc + j) * 264 + s] = vv[j];
    }
  }
  __syncthreads();

  int w = threadIdx.x >> 5, lane = threadIdx.x & 31;
  __bf16* Pw = Ps + w * 32 * 40;
  int q0 = w * 32;
  bf16x16 aq0 = load_Q(qb, st, cnt, h, q0, lane);        // Dh == WMMA K == 32
  bf16x16 aq1 = load_Q(qb, st, cnt, h, q0 + 16, lane);
  f32x8 m0v, l0v, m1v, l1v, o00 = {}, o01 = {}, o10 = {}, o11 = {};
#pragma unroll
  for (int i = 0; i < 8; ++i) { m0v[i] = -3.0e38f; m1v[i] = -3.0e38f; l0v[i] = 0.f; l1v[i] = 0.f; }
  for (int kt = 0; kt < 8; ++kt) {                       // 32-key tiles
    int key0 = kt * 32;
    bf16x16 bk0 = load_Bk(Ks, key0, lane);
    bf16x16 bk1 = load_Bk(Ks, key0 + 16, lane);
    f32x8 z = {};
    f32x8 s00 = wmma_bf16(aq0, bk0, z);
    f32x8 s01 = wmma_bf16(aq0, bk1, z);
    f32x8 s10 = wmma_bf16(aq1, bk0, z);
    f32x8 s11 = wmma_bf16(aq1, bk1, z);
    int keyA = key0 + (lane & 15), keyB = keyA + 16;     // key_padding_mask
    if (keyA >= cnt) {
#pragma unroll
      for (int i = 0; i < 8; ++i) { s00[i] = -1e9f; s10[i] = -1e9f; }
    }
    if (keyB >= cnt) {
#pragma unroll
      for (int i = 0; i < 8; ++i) { s01[i] = -1e9f; s11[i] = -1e9f; }
    }
    online_update(m0v, l0v, s00, s01, o00, o01, Pw, lane);
    online_update(m1v, l1v, s10, s11, o10, o11, Pw + 16 * 40, lane);
    asm volatile("s_wait_dscnt 0" ::: "memory");         // P LDS writes visible
    bf16x16 ap0 = load_A_lds(Pw, 40);
    bf16x16 ap1 = load_A_lds(Pw + 16 * 40, 40);
    bf16x16 bv0 = load_Bv(Vt, key0, 0, lane);
    bf16x16 bv1 = load_Bv(Vt, key0, 16, lane);
    o00 = wmma_bf16(ap0, bv0, o00);
    o01 = wmma_bf16(ap0, bv1, o01);
    o10 = wmma_bf16(ap1, bv0, o10);
    o11 = wmma_bf16(ap1, bv1, o11);
  }
  store_O(attn, st, cnt, h, q0, o00, o01, l0v, lane);
  store_O(attn, st, cnt, h, q0 + 16, o10, o11, l1v, lane);
}

// ================================================================ launcher
extern "C" void kernel_launch(void* const* d_in, const int* in_sizes, int n_in,
                              void* d_out, int out_size, void* d_ws, size_t ws_size,
                              hipStream_t stream) {
  const float* x     = (const float*)d_in[0];
  const int*   ei    = (const int*)d_in[1];
  const int*   batch = (const int*)d_in[2];
  const float* Wroot = (const float*)d_in[3];
  const float* Wnei  = (const float*)d_in[4];
  const float* bnei  = (const float*)d_in[5];
  const float* Wqkv  = (const float*)d_in[6];
  const float* bqkv  = (const float*)d_in[7];
  const float* Wo    = (const float*)d_in[8];
  const float* bo    = (const float*)d_in[9];
  const float* W1    = (const float*)d_in[10];
  const float* b1    = (const float*)d_in[11];
  const float* W2    = (const float*)d_in[12];
  const float* b2    = (const float*)d_in[13];
  const float* g1 = (const float*)d_in[14], *be1 = (const float*)d_in[15];
  const float* m1 = (const float*)d_in[16], *v1  = (const float*)d_in[17];
  const float* g2 = (const float*)d_in[18], *be2 = (const float*)d_in[19];
  const float* m2 = (const float*)d_in[20], *v2  = (const float*)d_in[21];
  const float* g3 = (const float*)d_in[22], *be3 = (const float*)d_in[23];
  const float* m3 = (const float*)d_in[24], *v3  = (const float*)d_in[25];
  (void)in_sizes; (void)n_in; (void)out_size; (void)ws_size;

  char* p = (char*)d_ws;
  auto take = [&](size_t bytes) -> void* {
    void* r = (void*)p;
    p += (bytes + 255) & ~size_t(255);
    return r;
  };
  float*  agg   = (float*)take((size_t)N_ * C_ * 4);
  float*  h1b   = (float*)take((size_t)N_ * C_ * 4);
  float*  outb  = (float*)take((size_t)N_ * C_ * 4);
  float*  attnb = (float*)take((size_t)N_ * C_ * 4);
  __bf16* qb    = (__bf16*)take((size_t)N_ * C_ * 2);
  __bf16* kbuf  = (__bf16*)take((size_t)N_ * C_ * 2);
  __bf16* vbuf  = (__bf16*)take((size_t)N_ * C_ * 2);
  __bf16* tbuf  = (__bf16*)take((size_t)N_ * 512 * 2);
  __bf16* wr    = (__bf16*)take(65536 * 2);
  __bf16* wn    = (__bf16*)take(65536 * 2);
  __bf16* wq    = (__bf16*)take(196608 * 2);
  __bf16* wo    = (__bf16*)take(65536 * 2);
  __bf16* w1b   = (__bf16*)take(131072 * 2);
  __bf16* w2b   = (__bf16*)take(131072 * 2);
  int*    counts = (int*)take(B_ * 4);
  int*    starts = (int*)take(B_ * 4);

  (void)hipMemsetAsync(agg, 0, (size_t)N_ * C_ * 4, stream);
  (void)hipMemsetAsync(counts, 0, B_ * 4, stream);

  k_cast<<<(65536 + 255) / 256, 256, 0, stream>>>(Wroot, wr, 65536);
  k_cast<<<(65536 + 255) / 256, 256, 0, stream>>>(Wnei, wn, 65536);
  k_cast<<<(196608 + 255) / 256, 256, 0, stream>>>(Wqkv, wq, 196608);
  k_cast<<<(65536 + 255) / 256, 256, 0, stream>>>(Wo, wo, 65536);
  k_cast<<<(131072 + 255) / 256, 256, 0, stream>>>(W1, w1b, 131072);
  k_cast<<<(131072 + 255) / 256, 256, 0, stream>>>(W2, w2b, 131072);

  k_count<<<(N_ + 255) / 256, 256, 0, stream>>>(batch, counts, N_);
  k_scan<<<1, 32, 0, stream>>>(counts, starts);

  k_edge<<<(E_ * 64) / 256, 256, 0, stream>>>(ei, x, agg, E_);

  k_conv<<<(N_ + 31) / 32, 256, 0, stream>>>(x, agg, wr, wn, bnei,
                                             g1, be1, m1, v1, h1b, N_);

  k_gemm<768, EpiQKV><<<(N_ + 63) / 64, 256, 0, stream>>>(
      x, wq, EpiQKV{bqkv, qb, kbuf, vbuf}, N_);

  k_attn<<<B_ * H_, 256, 0, stream>>>(qb, kbuf, vbuf, counts, starts, attnb);

  k_gemm<256, EpiOproj><<<(N_ + 63) / 64, 256, 0, stream>>>(
      attnb, wo, EpiOproj{bo, x, g2, be2, m2, v2, h1b, outb}, N_);

  k_gemm<512, EpiMlp1><<<(N_ + 63) / 64, 256, 0, stream>>>(
      outb, w1b, EpiMlp1{b1, tbuf}, N_);

  k_mlp2<<<(N_ + 63) / 64, 256, 0, stream>>>(tbuf, w2b, b2, outb,
                                             g3, be3, m3, v3, (float*)d_out, N_);
}
